// DeformingPlateModel_31645319037519
// MI455X (gfx1250) — matrix operand order repaired and checked by reading, not compiled, not run on hardware
//
#include <hip/hip_runtime.h>
#include <math.h>

#define NN 20000
#define EM 100000
#define EW 25000

typedef _Float16 h16;
typedef __attribute__((ext_vector_type(16))) _Float16 v16h;
typedef __attribute__((ext_vector_type(8)))  _Float16 v8h;
typedef __attribute__((ext_vector_type(8)))  float    v8f;
typedef __attribute__((ext_vector_type(4)))  float    v4f;

// ---------------------------------------------------------------------------
// Fully fused 3-layer MLP over a 32-row tile, single 24.9KB LDS arena:
//   X  = concat(seg0,seg1,seg2)           [32, Kh0]   arena[0 .. 32*Kh0*2)
//   H1 = relu(X @ W0 + b0)                [32, 128]   arena[0 .. 8K)    (aliased)
//   H2 = relu(H1 @ W1 + b1)               [32, 128]   arena[8K .. 16K)
//   Y  = H2 @ W2 + b2 (+LN(g,b)) (+res)   [32, NoutL] f32 LN tile arena[0..16K)
// 8 waves; each wave owns one 16-col tile and two 16-row subtiles, so every
// B-fragment load feeds two v_wmma_f32_16x16x32_f16.
// ---------------------------------------------------------------------------
__global__ __launch_bounds__(256) void mlp3_fused(
    int M, int Kh0, int NoutL, int nRealL,
    const void* s0p, const int* s0i, int s0w, int s0f,
    const void* s1p, const int* s1i, int s1w, int s1f,
    const void* s2p, const int* s2i, int s2w, int s2f,
    const h16* __restrict__ Wbase,
    const float* __restrict__ b0, const float* __restrict__ b1,
    const float* __restrict__ b2,
    const float* __restrict__ lng, const float* __restrict__ lnb,
    const h16* __restrict__ resid, h16* __restrict__ Y)
{
  __shared__ __align__(16) char smem[32 * 384 * 2 + 64 * 4];
  h16*   sX  = (h16*)smem;               // [0, 32*Kh0*2)
  h16*   sHa = (h16*)smem;               // [0, 8K)   layer-1 output
  h16*   sHb = (h16*)(smem + 8192);      // [8K, 16K) layer-2 output
  float* sD  = (float*)smem;             // [0, 16K)  f32 LN tile
  float* sMu = (float*)(smem + 32 * 384 * 2);
  float* sR  = sMu + 32;

  const int tid = threadIdx.x;
  const int m0  = blockIdx.x * 32;

  const h16* W1 = Wbase + (size_t)Kh0 * 128;
  const h16* W2 = W1 + 128 * 128;

  // warm near-cache with layer-2/3 weights while we stage/compute layer 1
  __builtin_prefetch(W1 + (size_t)tid * 64, 0, 0);
  __builtin_prefetch(W2 + (size_t)tid * 64, 0, 0);

  // ---- stage X tile: vectorized 8-half (16B) chunks, 3-segment gather
  const int KhV = Kh0 >> 3;
  for (int i = tid; i < 32 * KhV; i += 256) {
    int r = i / KhV, kv = i - r * KhV, k = kv << 3;
    int m = m0 + r;
    v8h val;
#pragma unroll
    for (int e = 0; e < 8; ++e) val[e] = (h16)0.f;
    if (m < M) {
      const void* p; const int* id; int w, f, kk;
      if (k < s0w)            { p = s0p; id = s0i; w = s0w; f = s0f; kk = k; }
      else if (k < s0w + s1w) { p = s1p; id = s1i; w = s1w; f = s1f; kk = k - s0w; }
      else                    { p = s2p; id = s2i; w = s2w; f = s2f; kk = k - s0w - s1w; }
      int sr = id ? id[m] : m;
      if (f) {
        const float* pf = (const float*)p + (size_t)sr * w + kk;
        v4f lo = *(const v4f*)pf;
        v4f hi = *(const v4f*)(pf + 4);
#pragma unroll
        for (int e = 0; e < 4; ++e) { val[e] = (h16)lo[e]; val[4 + e] = (h16)hi[e]; }
      } else {
        val = *(const v8h*)((const h16*)p + (size_t)sr * w + kk);
      }
    }
    *(v8h*)&sX[r * Kh0 + k] = val;
  }
  __syncthreads();

  const int wave = tid >> 5, lane = tid & 31;
  const int g = lane >> 4, nl = lane & 15;
  const int ncol = wave * 16 + nl;

  // one 16-col tile, two 16-row subtiles sharing each B fragment
  auto gemm2 = [&](const h16* src, int Kh, const h16* Wp, v8f& acc0, v8f& acc1) {
    const int nKc = Kh >> 5;
    const v16h* Wv = (const v16h*)Wp;
    for (int kc = 0; kc < nKc; ++kc) {
      v16h b = Wv[(wave * nKc + kc) * 32 + lane];
      const h16* Ar0 = &src[nl * Kh + kc * 32 + g * 16];
      const h16* Ar1 = Ar0 + 16 * Kh;
      v16h a0, a1;
#pragma unroll
      for (int e = 0; e < 8; ++e) {
        a0[e] = Ar0[e]; a0[e + 8] = Ar0[8 + e];
        a1[e] = Ar1[e]; a1[e + 8] = Ar1[8 + e];
      }
      acc0 = __builtin_amdgcn_wmma_f32_16x16x32_f16(false, a0, false, b,
                                                    (short)0, acc0, false, false);
      acc1 = __builtin_amdgcn_wmma_f32_16x16x32_f16(false, a1, false, b,
                                                    (short)0, acc1, false, false);
    }
  };

  const v8f vz = {0.f, 0.f, 0.f, 0.f, 0.f, 0.f, 0.f, 0.f};
  v8f c0, c1;
  float bb;

  // ---- layer 1: sX -> registers -> sHa (relu); sX dead after the sync
  c0 = vz; c1 = vz;
  gemm2(sX, Kh0, Wbase, c0, c1);
  __syncthreads();                     // everyone done reading sX
  bb = b0[ncol];
#pragma unroll
  for (int j = 0; j < 8; ++j) {
    float x0 = c0[j] + bb; x0 = x0 > 0.f ? x0 : 0.f;
    float x1 = c1[j] + bb; x1 = x1 > 0.f ? x1 : 0.f;
    sHa[(g * 8 + j) * 128 + ncol]      = (h16)x0;
    sHa[(16 + g * 8 + j) * 128 + ncol] = (h16)x1;
  }
  __syncthreads();

  // ---- layer 2: sHa -> sHb (relu)
  c0 = vz; c1 = vz;
  gemm2(sHa, 128, W1, c0, c1);
  bb = b1[ncol];
#pragma unroll
  for (int j = 0; j < 8; ++j) {
    float x0 = c0[j] + bb; x0 = x0 > 0.f ? x0 : 0.f;
    float x1 = c1[j] + bb; x1 = x1 > 0.f ? x1 : 0.f;
    sHb[(g * 8 + j) * 128 + ncol]      = (h16)x0;
    sHb[(16 + g * 8 + j) * 128 + ncol] = (h16)x1;
  }
  __syncthreads();

  // ---- layer 3: sHb -> Y  (bias; optional LN + residual)
  const bool act3 = (wave * 16) < NoutL;   // decoder: only wave 0
  c0 = vz; c1 = vz;
  if (act3) {
    gemm2(sHb, 128, W2, c0, c1);
    bb = (ncol < nRealL) ? b2[ncol] : 0.f;
#pragma unroll
    for (int j = 0; j < 8; ++j) { c0[j] += bb; c1[j] += bb; }
  }

  if (!lng) {
    if (act3) {
#pragma unroll
      for (int j = 0; j < 8; ++j) {
        int ma = m0 + g * 8 + j, mb = ma + 16;
        if (ma < M) Y[(size_t)ma * NoutL + ncol] = (h16)c0[j];
        if (mb < M) Y[(size_t)mb * NoutL + ncol] = (h16)c1[j];
      }
    }
  } else {
    __syncthreads();                   // everyone done reading sHa/sHb
    if (act3) {
#pragma unroll
      for (int j = 0; j < 8; ++j) {
        sD[(g * 8 + j) * 128 + ncol]      = c0[j];
        sD[(16 + g * 8 + j) * 128 + ncol] = c1[j];
      }
    }
    __syncthreads();
    if (tid < 32) {
      float s = 0.f, s2 = 0.f;
      for (int n = 0; n < 128; ++n) { float x = sD[tid * 128 + n]; s += x; s2 += x * x; }
      float mu = s * (1.f / 128.f);
      float var = s2 * (1.f / 128.f) - mu * mu;
      sMu[tid] = mu; sR[tid] = rsqrtf(var + 1e-5f);
    }
    __syncthreads();
    for (int i = tid; i < 32 * 128; i += 256) {
      int r = i >> 7, n = i & 127;
      int m = m0 + r;
      if (m < M) {
        float x = (sD[i] - sMu[r]) * sR[r] * lng[n] + lnb[n];
        if (resid) x += (float)resid[(size_t)m * 128 + n];
        Y[(size_t)m * 128 + n] = (h16)x;
      }
    }
  }
}

// ---------------------------------------------------------------------------
// Pack fp32 weight [Kreal,Nreal] -> f16 WMMA B-fragments [KhPad x Nout]
// out[((t*nKc+kc)*32+lane)*16+e], K = kc*32 + (lane>=16)*16 + e
// ---------------------------------------------------------------------------
__global__ void pack_w(const float* __restrict__ W, h16* __restrict__ out,
                       int Kreal, int Kh, int Nreal, int Nout, int total)
{
  int i = blockIdx.x * blockDim.x + threadIdx.x;
  if (i >= total) return;
  int e = i & 15, lane = (i >> 4) & 31, frag = i >> 9;
  int nKc = Kh >> 5;
  int kc = frag % nKc, t = frag / nKc;
  int k = kc * 32 + (lane >> 4) * 16 + e;
  int n = t * 16 + (lane & 15);
  float v = (k < Kreal && n < Nreal) ? W[(size_t)k * Nreal + n] : 0.f;
  out[i] = (h16)v;
}

// ---------------------------------------------------------------------------
__global__ void node_feat_k(const int* __restrict__ nt, const float* __restrict__ kv,
                            const float* __restrict__ mean, const float* __restrict__ stdv,
                            h16* __restrict__ out, int N)
{
  int i = blockIdx.x * blockDim.x + threadIdx.x;
  if (i >= N) return;
  int t = nt[i];
  for (int c = 0; c < 32; ++c) {
    float f = 0.f;
    if (c < 3)       f = kv[i * 3 + c];
    else if (c < 12) f = (t == c - 3) ? 1.f : 0.f;
    float o = (c < 12) ? (f - mean[c]) / stdv[c] : 0.f;
    out[(size_t)i * 32 + c] = (h16)o;
  }
}

__global__ void mesh_feat_k(const float* __restrict__ mp, const float* __restrict__ wp,
                            const int* __restrict__ s, const int* __restrict__ d,
                            const float* __restrict__ mean, const float* __restrict__ stdv,
                            h16* __restrict__ out, int E)
{
  int i = blockIdx.x * blockDim.x + threadIdx.x;
  if (i >= E) return;
  int a = s[i], b = d[i];
  float f[8];
  float r0 = mp[a*3+0] - mp[b*3+0];
  float r1 = mp[a*3+1] - mp[b*3+1];
  float r2 = mp[a*3+2] - mp[b*3+2];
  f[0]=r0; f[1]=r1; f[2]=r2; f[3]=sqrtf(r0*r0+r1*r1+r2*r2);
  float q0 = wp[a*3+0] - mp[b*3+0];
  float q1 = wp[a*3+1] - mp[b*3+1];
  float q2 = wp[a*3+2] - mp[b*3+2];
  f[4]=q0; f[5]=q1; f[6]=q2; f[7]=sqrtf(q0*q0+q1*q1+q2*q2);
  for (int c = 0; c < 32; ++c)
    out[(size_t)i*32 + c] = (c < 8) ? (h16)((f[c] - mean[c]) / stdv[c]) : (h16)0.f;
}

__global__ void world_feat_k(const float* __restrict__ wp,
                             const int* __restrict__ s, const int* __restrict__ d,
                             const float* __restrict__ mean, const float* __restrict__ stdv,
                             h16* __restrict__ out, int E)
{
  int i = blockIdx.x * blockDim.x + threadIdx.x;
  if (i >= E) return;
  int a = s[i], b = d[i];
  float f[4];
  float r0 = wp[a*3+0] - wp[b*3+0];
  float r1 = wp[a*3+1] - wp[b*3+1];
  float r2 = wp[a*3+2] - wp[b*3+2];
  f[0]=r0; f[1]=r1; f[2]=r2; f[3]=sqrtf(r0*r0+r1*r1+r2*r2);
  for (int c = 0; c < 32; ++c)
    out[(size_t)i*32 + c] = (c < 4) ? (h16)((f[c] - mean[c]) / stdv[c]) : (h16)0.f;
}

__global__ void zero_f32(float* p, int n)
{
  int i = blockIdx.x * blockDim.x + threadIdx.x;
  if (i < n) p[i] = 0.f;
}

__global__ void scatter_add_k(const h16* __restrict__ e, const int* __restrict__ dst,
                              float* __restrict__ agg, int E)
{
  int i = blockIdx.x * blockDim.x + threadIdx.x;
  if (i >= E * 128) return;
  int ed = i >> 7, c = i & 127;
  atomicAdd(&agg[(size_t)dst[ed] * 128 + c], (float)e[i]);
}

__global__ void unnorm_k(const h16* __restrict__ y, const float* __restrict__ mean,
                         const float* __restrict__ stdv, float* __restrict__ out, int N)
{
  int i = blockIdx.x * blockDim.x + threadIdx.x;
  if (i >= N * 3) return;
  int r = i / 3, c = i - r * 3;
  out[i] = (float)y[(size_t)r * 16 + c] * stdv[c] + mean[c];
}

// ---------------------------------------------------------------------------
extern "C" void kernel_launch(void* const* d_in, const int* in_sizes, int n_in,
                              void* d_out, int out_size, void* d_ws, size_t ws_size,
                              hipStream_t stream)
{
  (void)in_sizes; (void)n_in; (void)out_size; (void)ws_size;

  // ---- inputs (setup_inputs dict order; params flattened in insertion order)
  const int*   node_type = (const int*)  d_in[0];
  const float* mesh_pos  = (const float*)d_in[1];
  const float* world_pos = (const float*)d_in[2];
  const float* known_vel = (const float*)d_in[3];
  const int*   srcs      = (const int*)  d_in[4];
  const int*   dsts      = (const int*)  d_in[5];
  const int*   wsrcs     = (const int*)  d_in[6];
  const int*   wdsts     = (const int*)  d_in[7];

  int P = 8;
  auto nf32 = [&]() { return (const float*)d_in[P++]; };

  struct Mlp { const float *W0,*b0,*W1,*b1,*W2,*b2,*g,*bt; };
  auto readMlp = [&](bool ln) {
    Mlp m;
    m.W0 = nf32(); m.b0 = nf32();
    m.W1 = nf32(); m.b1 = nf32();
    m.W2 = nf32(); m.b2 = nf32();
    if (ln) { m.g = nf32(); m.bt = nf32(); } else { m.g = nullptr; m.bt = nullptr; }
    return m;
  };

  Mlp node_enc  = readMlp(true);
  Mlp mesh_enc  = readMlp(true);
  Mlp world_enc = readMlp(true);
  Mlp blk[15][3];
  for (int i = 0; i < 15; ++i) {
    blk[i][0] = readMlp(true);   // mesh edge
    blk[i][1] = readMlp(true);   // world edge
    blk[i][2] = readMlp(true);   // node
  }
  Mlp dec = readMlp(false);
  const float* nmean = nf32(); const float* nstd = nf32();
  const float* mmean = nf32(); const float* mstd = nf32();
  const float* wmean = nf32(); const float* wstd = nf32();
  const float* omean = nf32(); const float* ostd = nf32();

  // ---- workspace carve-up
  char* wc = (char*)d_ws;
  auto alloc = [&](size_t bytes) -> void* {
    void* p = wc; wc += (bytes + 255) & ~(size_t)255; return p;
  };
  h16*   v    = (h16*)  alloc((size_t)NN * 128 * 2);
  h16*   em   = (h16*)  alloc((size_t)EM * 128 * 2);
  h16*   ew   = (h16*)  alloc((size_t)EW * 128 * 2);
  float* aggm = (float*)alloc((size_t)NN * 128 * 4);
  float* aggw = (float*)alloc((size_t)NN * 128 * 4);
  h16*   nfb  = (h16*)  alloc((size_t)NN * 32 * 2);
  h16*   mfb  = (h16*)  alloc((size_t)EM * 32 * 2);
  h16*   wfb  = (h16*)  alloc((size_t)EW * 32 * 2);
  h16*   dect = (h16*)  alloc((size_t)NN * 16 * 2);
  h16*   Wpk  = (h16*)  alloc((size_t)4000000 * 2);

  // ---- weight packing (w0,w1,w2 of each MLP stay contiguous)
  size_t wcur = 0;
  auto pack = [&](const float* W, int Kreal, int Nreal, int KhPad, int Nout) -> const h16* {
    h16* dst = Wpk + wcur;
    int total = KhPad * Nout;
    wcur += (size_t)total;
    pack_w<<<(total + 255) / 256, 256, 0, stream>>>(W, dst, Kreal, KhPad, Nreal, Nout, total);
    return dst;
  };
  struct PMlp { const h16* w0; };
  auto packMlp = [&](const Mlp& m, int in, int outReal, int outPad) {
    PMlp p;
    int K0 = (in + 31) & ~31;
    p.w0 = pack(m.W0, in, 128, K0, 128);
    pack(m.W1, 128, 128, 128, 128);
    pack(m.W2, 128, outReal, 128, outPad);
    return p;
  };
  PMlp p_node  = packMlp(node_enc, 12, 128, 128);
  PMlp p_mesh  = packMlp(mesh_enc, 8, 128, 128);
  PMlp p_world = packMlp(world_enc, 4, 128, 128);
  PMlp p_blk[15][3];
  for (int i = 0; i < 15; ++i)
    for (int j = 0; j < 3; ++j)
      p_blk[i][j] = packMlp(blk[i][j], 384, 128, 128);
  PMlp p_dec = packMlp(dec, 128, 3, 16);

  // ---- fused MLP launch helper
  struct SegH { const void* p; const int* i; int w; int f; };
  SegH Z{nullptr, nullptr, 0, 0};
  auto runMlp = [&](const Mlp& m, const PMlp& p, int M, int Kh0,
                    SegH s0, SegH s1, SegH s2,
                    h16* Y, const h16* resid, bool ln, int NoutL, int nRealL) {
    dim3 grid((M + 31) / 32), tb(256);
    mlp3_fused<<<grid, tb, 0, stream>>>(M, Kh0, NoutL, nRealL,
        s0.p, s0.i, s0.w, s0.f, s1.p, s1.i, s1.w, s1.f, s2.p, s2.i, s2.w, s2.f,
        p.w0, m.b0, m.b1, m.b2,
        ln ? m.g : nullptr, ln ? m.bt : nullptr, resid, Y);
  };

  // ---- features
  node_feat_k<<<(NN + 255) / 256, 256, 0, stream>>>(node_type, known_vel, nmean, nstd, nfb, NN);
  mesh_feat_k<<<(EM + 255) / 256, 256, 0, stream>>>(mesh_pos, world_pos, srcs, dsts, mmean, mstd, mfb, EM);
  world_feat_k<<<(EW + 255) / 256, 256, 0, stream>>>(world_pos, wsrcs, wdsts, wmean, wstd, wfb, EW);

  // ---- encoders
  runMlp(node_enc,  p_node,  NN, 32, SegH{nfb, nullptr, 32, 0}, Z, Z, v,  nullptr, true, 128, 128);
  runMlp(mesh_enc,  p_mesh,  EM, 32, SegH{mfb, nullptr, 32, 0}, Z, Z, em, nullptr, true, 128, 128);
  runMlp(world_enc, p_world, EW, 32, SegH{wfb, nullptr, 32, 0}, Z, Z, ew, nullptr, true, 128, 128);

  // ---- 15 residual message-passing blocks
  for (int i = 0; i < 15; ++i) {
    // mesh edge update: e = MLP([v[s], v[d], e]) + e
    runMlp(blk[i][0], p_blk[i][0], EM, 384,
           SegH{v, srcs, 128, 0}, SegH{v, dsts, 128, 0}, SegH{em, nullptr, 128, 0},
           em, em, true, 128, 128);
    // world edge update
    runMlp(blk[i][1], p_blk[i][1], EW, 384,
           SegH{v, wsrcs, 128, 0}, SegH{v, wdsts, 128, 0}, SegH{ew, nullptr, 128, 0},
           ew, ew, true, 128, 128);
    // aggregate (segment-sum over destinations)
    zero_f32<<<(NN * 128 + 255) / 256, 256, 0, stream>>>(aggm, NN * 128);
    zero_f32<<<(NN * 128 + 255) / 256, 256, 0, stream>>>(aggw, NN * 128);
    scatter_add_k<<<(EM * 128 + 255) / 256, 256, 0, stream>>>(em, dsts, aggm, EM);
    scatter_add_k<<<(EW * 128 + 255) / 256, 256, 0, stream>>>(ew, wdsts, aggw, EW);
    // node update: v = MLP([v, aggm, aggw]) + v
    runMlp(blk[i][2], p_blk[i][2], NN, 384,
           SegH{v, nullptr, 128, 0}, SegH{aggm, nullptr, 128, 1}, SegH{aggw, nullptr, 128, 1},
           v, v, true, 128, 128);
  }

  // ---- decode + unnorm
  runMlp(dec, p_dec, NN, 128, SegH{v, nullptr, 128, 0}, Z, Z, dect, nullptr, false, 16, 3);
  unnorm_k<<<(NN * 3 + 255) / 256, 256, 0, stream>>>(dect, omean, ostd, (float*)d_out, NN);
}
